// MultiheadSlidingWindowSelfAttention_37374805410024
// MI455X (gfx1250) — matrix operand order repaired and checked by reading, not compiled, not run on hardware
//
#include <hip/hip_runtime.h>

// ---------------------------------------------------------------------------
// MI455X (gfx1250) sliding-window MHA.
//   hidden f32 [4,4096,1024] -> QKV proj (bf16 WMMA, f32 accum)
//   -> chunked sliding-window flash attention (bf16 WMMA)
//   -> output projection -> f32 out.
// Workspace: Q,K,V [B,H,T,D] bf16 + attn-out [B*T,E] bf16 = 4 * 32MB = 128MB.
// ---------------------------------------------------------------------------

typedef __attribute__((ext_vector_type(16))) __bf16 v16bf;
typedef __attribute__((ext_vector_type(8)))  __bf16 v8bf;
typedef __attribute__((ext_vector_type(8)))  float  v8f;

#define SEQ_T   4096
#define EMBED_E 1024
#define NHEADS  16
#define HDIM    64
#define CHUNK_C 256
#define NCHUNK  16
#define BATCH_B 4
#define MROWS   (BATCH_B * SEQ_T)        // 16384
#define NEG_INF (-__builtin_inff())

__device__ __forceinline__ v16bf pack16(v8bf lo, v8bf hi) {
  v16bf r;
#pragma unroll
  for (int i = 0; i < 8; ++i) { r[i] = lo[i]; r[i + 8] = hi[i]; }
  return r;
}

__device__ __forceinline__ v8f v8f_zero() {
  v8f z;
#pragma unroll
  for (int i = 0; i < 8; ++i) z[i] = 0.0f;
  return z;
}

__device__ __forceinline__ v8bf v8bf_zero() {
  v8bf z;
#pragma unroll
  for (int i = 0; i < 8; ++i) z[i] = (__bf16)0.0f;
  return z;
}

__device__ __forceinline__ v8f wmma_bf16(v16bf a, v16bf b, v8f c) {
  // D = A(16x32 bf16) * B(32x16 bf16) + C(16x16 f32)
  return __builtin_amdgcn_wmma_f32_16x16x32_bf16(false, a, false, b,
                                                 (short)0, c, false, false);
}

// ---------------------------------------------------------------------------
// Generic M=16384, N=1024, K=1024 GEMM:  Out = scale * (A @ W + bias)
//   A_F32:    A is f32 (hidden states) else bf16 (attention output)
//   SPLIT:    scatter output to [B,H,T,D] bf16 (QKV projections)
//   (else)    dense f32 [M,N] output (final projection)
// Block 256 threads = 8 waves (4 M x 2 N), tile 128x128, BK=32.
// ---------------------------------------------------------------------------
template <bool A_F32, bool SPLIT>
__global__ __launch_bounds__(256)
void gemm_bias_kernel(const void* __restrict__ Aptr,
                      const float* __restrict__ W,
                      const float* __restrict__ bias,
                      void* __restrict__ Out,
                      float scale) {
  constexpr int BM = 128, BN = 128, BK = 32;
  constexpr int LDA = BK + 8;                 // 40 elems -> 80B row, 16B aligned
  __shared__ __bf16 As[BM][LDA];              // [m][k]
  __shared__ __bf16 Bs[BN][LDA];              // transposed: [n][k]

  const int tid  = threadIdx.x;
  const int wave = tid >> 5;
  const int lane = tid & 31;
  const int ln   = lane & 15;
  const int lh   = lane >> 4;

  const int bm = blockIdx.y * BM;
  const int bn = blockIdx.x * BN;
  const int waveM = (wave >> 1) * 32;
  const int waveN = (wave & 1) * 64;

  const float*  Af = (const float*)Aptr;
  const __bf16* Ab = (const __bf16*)Aptr;

  v8f acc[2][4];
#pragma unroll
  for (int i = 0; i < 2; ++i)
#pragma unroll
    for (int j = 0; j < 4; ++j) acc[i][j] = v8f_zero();

  for (int kb = 0; kb < EMBED_E; kb += BK) {
    __syncthreads();
    // ---- stage A tile 128x32 -> bf16 LDS -------------------------------
    {
      const int row = tid >> 1;
      const int c0  = (tid & 1) * 16;
      if (A_F32) {
#pragma unroll
        for (int j = 0; j < 4; ++j) {
          const float4 v =
              *(const float4*)(Af + (size_t)(bm + row) * EMBED_E + kb + c0 + j * 4);
          As[row][c0 + j * 4 + 0] = (__bf16)v.x;
          As[row][c0 + j * 4 + 1] = (__bf16)v.y;
          As[row][c0 + j * 4 + 2] = (__bf16)v.z;
          As[row][c0 + j * 4 + 3] = (__bf16)v.w;
        }
      } else {
        const __bf16* p = Ab + (size_t)(bm + row) * EMBED_E + kb + c0;
        const v8bf lo = *(const v8bf*)p;
        const v8bf hi = *(const v8bf*)(p + 8);
#pragma unroll
        for (int j = 0; j < 8; ++j) {
          As[row][c0 + j]     = lo[j];
          As[row][c0 + 8 + j] = hi[j];
        }
      }
    }
    // ---- stage W tile 32x128 transposed -> Bs[n][k] --------------------
    {
      const int k  = tid >> 3;
      const int n0 = (tid & 7) * 16;
#pragma unroll
      for (int j = 0; j < 4; ++j) {
        const float4 v =
            *(const float4*)(W + (size_t)(kb + k) * EMBED_E + bn + n0 + j * 4);
        Bs[n0 + j * 4 + 0][k] = (__bf16)v.x;
        Bs[n0 + j * 4 + 1][k] = (__bf16)v.y;
        Bs[n0 + j * 4 + 2][k] = (__bf16)v.z;
        Bs[n0 + j * 4 + 3][k] = (__bf16)v.w;
      }
    }
    __syncthreads();

    if (kb + BK < EMBED_E) {  // global_prefetch_b8 for next A block
      const size_t off = (size_t)(bm + (tid >> 1)) * EMBED_E + kb + BK;
      const char* nxt = A_F32 ? (const char*)(Af + off) : (const char*)(Ab + off);
      __builtin_prefetch(nxt, 0, 1);
    }

    // ---- preload all fragments, then batch the 8 WMMAs -----------------
    v16bf af[2];
#pragma unroll
    for (int mt = 0; mt < 2; ++mt) {
      // A frag: lane row = ln; elems 0..7 -> k=lh*8+0..7, 8..15 -> k=lh*8+16..23
      const __bf16* p = &As[waveM + mt * 16 + ln][lh * 8];
      af[mt] = pack16(*(const v8bf*)p, *(const v8bf*)(p + 16));
    }
    v16bf bf[4];
#pragma unroll
    for (int nt = 0; nt < 4; ++nt) {
      // B frag: lane col = ln; elems 0..7 -> k=lh*8+0..7, 8..15 -> k=lh*8+16..23
      const __bf16* p = &Bs[waveN + nt * 16 + ln][lh * 8];
      bf[nt] = pack16(*(const v8bf*)p, *(const v8bf*)(p + 16));
    }
#pragma unroll
    for (int nt = 0; nt < 4; ++nt)
#pragma unroll
      for (int mt = 0; mt < 2; ++mt)
        acc[mt][nt] = wmma_bf16(af[mt], bf[nt], acc[mt][nt]);
  }

  // ---- epilogue: C layout row = r + lh*8, col = ln ---------------------
#pragma unroll
  for (int nt = 0; nt < 4; ++nt) {
    const int col = bn + waveN + nt * 16 + ln;
    const float bv = bias[col];
#pragma unroll
    for (int mt = 0; mt < 2; ++mt) {
#pragma unroll
      for (int r = 0; r < 8; ++r) {
        const int row = bm + waveM + mt * 16 + lh * 8 + r;
        const float v = (acc[mt][nt][r] + bv) * scale;
        if (SPLIT) {  // [B,H,T,D] bf16 (1/sqrt(D) folded into Q's scale)
          const int b = row >> 12;
          const int t = row & (SEQ_T - 1);
          const int h = col >> 6;
          const int d = col & 63;
          ((__bf16*)Out)[((size_t)((b * NHEADS + h) * SEQ_T + t)) * HDIM + d] =
              (__bf16)v;
        } else {      // dense f32 [M,N]
          ((float*)Out)[(size_t)row * EMBED_E + col] = v;
        }
      }
    }
  }
}

// ---------------------------------------------------------------------------
// Chunked sliding-window attention. 1 block per (chunk, head, batch).
// 8 waves x 32 query rows; streams 24 tiles of 32 keys over the 768-key band.
// Online (flash) softmax in f32; QK^T and P@V on the bf16 WMMA pipe.
// ---------------------------------------------------------------------------
__global__ __launch_bounds__(256)
void swa_attention_kernel(const __bf16* __restrict__ Q,
                          const __bf16* __restrict__ K,
                          const __bf16* __restrict__ V,
                          __bf16* __restrict__ O) {
  __shared__ __bf16 Ks[32][64];      // [key][dim]
  __shared__ __bf16 VT[64][40];      // [dim][key] (transposed, padded)
  __shared__ __bf16 Ps[8][32][32];   // per-wave P staging [row][key]

  const int tid  = threadIdx.x;
  const int wave = tid >> 5;
  const int lane = tid & 31;
  const int ln   = lane & 15;
  const int lh   = lane >> 4;

  const int chunk = blockIdx.x;
  const int head  = blockIdx.y;
  const int batch = blockIdx.z;

  const size_t bh = ((size_t)batch * NHEADS + head) * SEQ_T * HDIM;
  const __bf16* qb  = Q + bh;
  const __bf16* kbp = K + bh;
  const __bf16* vbp = V + bh;

  const int qchunk0 = chunk * CHUNK_C;
  const int rowbase = wave * 32;          // within chunk
  const int kpos0   = qchunk0 - CHUNK_C;  // global pos of band key 0

  // ---- load Q fragments (held in registers for all 24 iterations) ------
  v16bf qf[2][2];
#pragma unroll
  for (int mt = 0; mt < 2; ++mt)
#pragma unroll
    for (int ks = 0; ks < 2; ++ks) {
      const __bf16* p =
          qb + (size_t)(qchunk0 + rowbase + mt * 16 + ln) * HDIM + ks * 32 + lh * 8;
      qf[mt][ks] = pack16(*(const v8bf*)p, *(const v8bf*)(p + 16));
    }

  v8f o[2][4];
  float mrow[2][8], lrow[2][8];
#pragma unroll
  for (int mt = 0; mt < 2; ++mt) {
#pragma unroll
    for (int dt = 0; dt < 4; ++dt) o[mt][dt] = v8f_zero();
#pragma unroll
    for (int r = 0; r < 8; ++r) { mrow[mt][r] = NEG_INF; lrow[mt][r] = 0.0f; }
  }

  for (int it = 0; it < 24; ++it) {
    const int kstart = kpos0 + it * 32;
    __syncthreads();
    {  // stage K tile 32x64 (zero-fill out-of-sequence keys)
      const int key = tid >> 3;
      const int dp  = (tid & 7) * 8;
      const int kp  = kstart + key;
      v8bf val = v8bf_zero();
      if (kp >= 0 && kp < SEQ_T) val = *(const v8bf*)(kbp + (size_t)kp * HDIM + dp);
      *(v8bf*)&Ks[key][dp] = val;
    }
    {  // stage V tile transposed -> VT[dim][key]
      const int key = tid & 31;
      const int db  = (tid >> 5) * 8;
      const int kp  = kstart + key;
      v8bf val = v8bf_zero();
      if (kp >= 0 && kp < SEQ_T) val = *(const v8bf*)(vbp + (size_t)kp * HDIM + db);
#pragma unroll
      for (int j = 0; j < 8; ++j) VT[db + j][key] = val[j];
    }
    __syncthreads();

    // ---- phase 1: preload K frags (shared by both row tiles), batch the
    //      8 score WMMAs ------------------------------------------------
    v16bf kf[2][2];
#pragma unroll
    for (int ct = 0; ct < 2; ++ct)
#pragma unroll
      for (int ks = 0; ks < 2; ++ks) {
        const __bf16* p = &Ks[ct * 16 + ln][ks * 32 + lh * 8];
        kf[ct][ks] = pack16(*(const v8bf*)p, *(const v8bf*)(p + 16));
      }
    v8f s[2][2];  // [mt][ct]
#pragma unroll
    for (int mt = 0; mt < 2; ++mt)
#pragma unroll
      for (int ct = 0; ct < 2; ++ct) {
        v8f t = v8f_zero();
        t = wmma_bf16(qf[mt][0], kf[ct][0], t);
        t = wmma_bf16(qf[mt][1], kf[ct][1], t);
        s[mt][ct] = t;
      }

    // ---- phase 2: mask + online softmax per row tile -------------------
#pragma unroll
    for (int mt = 0; mt < 2; ++mt) {
      const int qbase = rowbase + mt * 16 + lh * 8;  // + r = query row in chunk
#pragma unroll
      for (int ct = 0; ct < 2; ++ct) {
        const int kj = it * 32 + ct * 16 + ln;  // band-local key index
        const int kp = kpos0 + kj;              // global key position
        const bool colok = (kp >= 0) && (kp < SEQ_T);
        const int d0 = kj - qbase;              // band cond: 0 <= d0-r <= 512
#pragma unroll
        for (int r = 0; r < 8; ++r) {
          const bool ok = colok && ((unsigned)(d0 - r) <= 2u * CHUNK_C);
          s[mt][ct][r] = ok ? s[mt][ct][r] : NEG_INF;
        }
      }
#pragma unroll
      for (int r = 0; r < 8; ++r) {
        float mx = fmaxf(s[mt][0][r], s[mt][1][r]);
#pragma unroll
        for (int off = 1; off <= 8; off <<= 1)
          mx = fmaxf(mx, __shfl_xor(mx, off, 32));
        const float mprev = mrow[mt][r];
        const float mnew  = fmaxf(mprev, mx);
        const float alpha = (mprev == NEG_INF) ? 0.0f : __expf(mprev - mnew);
        const float p0 = (s[mt][0][r] == NEG_INF) ? 0.0f : __expf(s[mt][0][r] - mnew);
        const float p1 = (s[mt][1][r] == NEG_INF) ? 0.0f : __expf(s[mt][1][r] - mnew);
        s[mt][0][r] = p0;
        s[mt][1][r] = p1;
        float rs = p0 + p1;
#pragma unroll
        for (int off = 1; off <= 8; off <<= 1) rs += __shfl_xor(rs, off, 32);
        lrow[mt][r] = lrow[mt][r] * alpha + rs;
        mrow[mt][r] = mnew;
#pragma unroll
        for (int dt = 0; dt < 4; ++dt) o[mt][dt][r] *= alpha;
      }
      // C-layout -> A-fragment staging through wave-private LDS
#pragma unroll
      for (int ct = 0; ct < 2; ++ct)
#pragma unroll
        for (int r = 0; r < 8; ++r)
          Ps[wave][mt * 16 + lh * 8 + r][ct * 16 + ln] = (__bf16)s[mt][ct][r];
    }

    // ---- phase 3: preload P and V frags, batch the 8 P@V WMMAs ---------
    v16bf pa[2];
#pragma unroll
    for (int mt = 0; mt < 2; ++mt) {
      const __bf16* pp = &Ps[wave][mt * 16 + ln][lh * 8];
      pa[mt] = pack16(*(const v8bf*)pp, *(const v8bf*)(pp + 16));
    }
    v16bf vf[4];
#pragma unroll
    for (int dt = 0; dt < 4; ++dt) {
      const __bf16* vp = &VT[dt * 16 + ln][lh * 8];
      vf[dt] = pack16(*(const v8bf*)vp, *(const v8bf*)(vp + 16));
    }
#pragma unroll
    for (int dt = 0; dt < 4; ++dt)
#pragma unroll
      for (int mt = 0; mt < 2; ++mt)
        o[mt][dt] = wmma_bf16(pa[mt], vf[dt], o[mt][dt]);
  }

  // ---- finalize: O /= l, store bf16 to [B*T, E] (head-sliced) -----------
#pragma unroll
  for (int mt = 0; mt < 2; ++mt) {
#pragma unroll
    for (int r = 0; r < 8; ++r) {
      const float inv = (lrow[mt][r] > 0.0f) ? 1.0f / lrow[mt][r] : 0.0f;
      const int t = qchunk0 + rowbase + mt * 16 + lh * 8 + r;
#pragma unroll
      for (int dt = 0; dt < 4; ++dt) {
        const float v = o[mt][dt][r] * inv;
        O[((size_t)(batch * SEQ_T + t)) * EMBED_E + head * HDIM + dt * 16 + ln] =
            (__bf16)v;
      }
    }
  }
}

// ---------------------------------------------------------------------------
extern "C" void kernel_launch(void* const* d_in, const int* in_sizes, int n_in,
                              void* d_out, int out_size, void* d_ws, size_t ws_size,
                              hipStream_t stream) {
  const float* hidden = (const float*)d_in[0];
  const float* wq = (const float*)d_in[1];
  const float* bq = (const float*)d_in[2];
  const float* wk = (const float*)d_in[3];
  const float* bk = (const float*)d_in[4];
  const float* wv = (const float*)d_in[5];
  const float* bv = (const float*)d_in[6];
  const float* wo = (const float*)d_in[7];
  const float* bo = (const float*)d_in[8];
  float* out = (float*)d_out;

  // Workspace layout (bf16): Q, K, V in [B,H,T,D]; attn-out in [B*T, E].
  const size_t NELT = (size_t)MROWS * EMBED_E;  // 16,777,216
  __bf16* qws = (__bf16*)d_ws;
  __bf16* kws = qws + NELT;
  __bf16* vws = kws + NELT;
  __bf16* aws = vws + NELT;

  const dim3 ggrid(EMBED_E / 128, MROWS / 128);  // (8, 128)

  // QKV projections (scale 1/sqrt(64)=0.125 folded into Q)
  gemm_bias_kernel<true, true><<<ggrid, 256, 0, stream>>>(hidden, wq, bq, qws, 0.125f);
  gemm_bias_kernel<true, true><<<ggrid, 256, 0, stream>>>(hidden, wk, bk, kws, 1.0f);
  gemm_bias_kernel<true, true><<<ggrid, 256, 0, stream>>>(hidden, wv, bv, vws, 1.0f);

  // Sliding-window attention
  swa_attention_kernel<<<dim3(NCHUNK, NHEADS, BATCH_B), 256, 0, stream>>>(
      qws, kws, vws, aws);

  // Output projection -> f32
  gemm_bias_kernel<false, false><<<ggrid, 256, 0, stream>>>(aws, wo, bo, out, 1.0f);
}
